// KMeanReservoir_53171695125221
// MI455X (gfx1250) — matrix operand order repaired and checked by reading, not compiled, not run on hardware
//
#include <hip/hip_runtime.h>
#include <hip/hip_bf16.h>

// ---------------------------------------------------------------------------
// Nearest-centroid assignment for MI455X (gfx1250, wave32, WMMA + TDM).
//   sim[n,k] = 2*x[n].c[k] - |x[n]|^2 - |c[k]|^2 ; outputs (max_sim, argmax).
// bf16x3 split-precision GEMM on v_wmma_f32_16x16x32_bf16; centroid tiles
// staged into LDS by the Tensor Data Mover (double-buffered, TENSORcnt sync)
// and shared by all 8 waves of the block; fused max/argmax in the WMMA C/D
// register layout.
// ---------------------------------------------------------------------------

typedef __attribute__((ext_vector_type(16))) __bf16    v16bf;
typedef __attribute__((ext_vector_type(8)))  float     v8f;
typedef __attribute__((ext_vector_type(8)))  unsigned  v8u;
typedef __attribute__((ext_vector_type(4)))  unsigned  v4u;
typedef __attribute__((ext_vector_type(8)))  int       v8i;
typedef __attribute__((ext_vector_type(4)))  int       v4i;

#define DIM      256                 // feature dimension
#define KC       1024                // number of centroids
#define NTILES   (KC / 16)           // 64 column tiles
#define TILE_DW  4096                // dwords per staged tile (hi 2048 + lo 2048)
#define NEG_INF  (-3.402823466e38f)

static __device__ __forceinline__ unsigned f2bf_rne(float f) {
    // fp32 -> bf16 bits, round-to-nearest-even
    unsigned u = __float_as_uint(f);
    return ((u + 0x7FFFu + ((u >> 16) & 1u)) >> 16) & 0xFFFFu;
}

static __device__ __forceinline__ float bf_bits_to_f(unsigned h) {
    return __uint_as_float(h << 16);
}

static __device__ __forceinline__ v16bf as_v16bf(v8u u) {
    return __builtin_bit_cast(v16bf, u);
}

// ---------------------------------------------------------------------------
// TDM: one descriptor moves one 16KB centroid tile (4096 dwords) into LDS.
// D# group0: count=1, lds_addr, global_addr[56:0], type=2.
// D# group1: data_size=4B, tensor_dim0=4096, tensor_dim1=1, tile_dim0=4096,
//            tile_dim1=1, dim0_stride=4096. Groups 2/3 unused (zero).
// Toolchain: 6-arg builtin (clang-23 / therock-10.0):
//   (uint32x4 g0, int32x8 g1, int32x4, int32x4, int32x8, i32 cpol)
// ---------------------------------------------------------------------------
static __device__ __forceinline__ void tdm_load_tile(const unsigned* gsrc,
                                                     unsigned lds_off) {
    unsigned long long ga = (unsigned long long)(uintptr_t)gsrc;
    v4u g0;
    g0[0] = 1u;                                                // count=1, user
    g0[1] = lds_off;                                           // lds_addr bytes
    g0[2] = (unsigned)(ga & 0xFFFFFFFFu);                      // gaddr[31:0]
    g0[3] = (unsigned)((ga >> 32) & 0x01FFFFFFu) | (2u << 30); // gaddr[56:32]|type=2
    v8i g1;
    g1[0] = (int)(2u << 16);                 // data_size=4B; no wg_mask/pad/iter
    g1[1] = (int)((TILE_DW & 0xFFFFu) << 16);        // tensor_dim0[15:0]
    g1[2] = (int)((TILE_DW >> 16) | (1u << 16));     // tensor_dim0[31:16]|dim1 lo
    g1[3] = (int)((unsigned)TILE_DW << 16);          // tile_dim0
    g1[4] = 1;                                       // tile_dim1=1, tile_dim2=0
    g1[5] = TILE_DW;                                 // tensor_dim0_stride lo
    g1[6] = (int)((TILE_DW & 0xFFFFu) << 16);        // dim1_stride lo (unused)
    g1[7] = 0;
    v4i gz4 = {0, 0, 0, 0};
    v8i gz8 = {0, 0, 0, 0, 0, 0, 0, 0};
    __builtin_amdgcn_tensor_load_to_lds(g0, g1, gz4, gz4, gz8, 0);
}

// ---------------------------------------------------------------------------
// Kernel 1: centroids fp32[KC][DIM] -> TDM-tiled bf16 hi/lo staging + |c|^2.
// Tile t (16 centroids): stage[t*4096 + r*128 + dw]        = hi pair dwords
//                        stage[t*4096 + 2048 + r*128 + dw] = lo pair dwords
// One wave per centroid row; lane handles dwords lane*4 .. lane*4+3.
// ---------------------------------------------------------------------------
__global__ __launch_bounds__(256)
void centroid_prep_kernel(const float* __restrict__ cen,
                          unsigned* __restrict__ stage,
                          float* __restrict__ cnorm) {
    const int lane = threadIdx.x & 31;
    const int wave = threadIdx.x >> 5;
    const int row  = blockIdx.x * 8 + wave;           // grid = KC/8 blocks

    const float* cr = cen + (size_t)row * DIM;
    const int t = row >> 4, r = row & 15;
    unsigned* hr = stage + (size_t)t * TILE_DW + r * (DIM / 2);
    unsigned* lr = hr + (TILE_DW / 2);

    float sum = 0.0f;
#pragma unroll
    for (int j = 0; j < 4; ++j) {
        const int dw = lane * 4 + j;                  // dword index in row
        const float f0 = cr[dw * 2 + 0];
        const float f1 = cr[dw * 2 + 1];
        const unsigned h0 = f2bf_rne(f0), h1 = f2bf_rne(f1);
        const unsigned l0 = f2bf_rne(f0 - bf_bits_to_f(h0));
        const unsigned l1 = f2bf_rne(f1 - bf_bits_to_f(h1));
        hr[dw] = h0 | (h1 << 16);
        lr[dw] = l0 | (l1 << 16);
        sum += f0 * f0 + f1 * f1;
    }
#pragma unroll
    for (int off = 16; off >= 1; off >>= 1)
        sum += __shfl_xor(sum, off);
    if (lane == 0) cnorm[row] = sum;
}

// ---------------------------------------------------------------------------
// Kernel 2: fused WMMA GEMM + argmax, TDM-staged B tiles in LDS.
//   A (16x32 bf16) layout: lane L -> M=L%16; half=L/16 selects K-groups
//     {half*16..+15} of each 32-wide d-step (element e holds K=half*16+e).
//   B (32x16 bf16) layout: lane L -> N=L%16; VGPR j holds K=half*16+2j,+1.
//   C/D (16x16 f32) layout: VGPR i -> row i+8*half, column L%16.
// ---------------------------------------------------------------------------
__global__ __launch_bounds__(256)
void nearest_centroid_kernel(const float* __restrict__ x,
                             const unsigned* __restrict__ stage, // [64][4096]
                             const float* __restrict__ cnorm,    // [KC]
                             float* __restrict__ out_sim,
                             int*   __restrict__ out_idx) {
    __shared__ unsigned ldsb[2][TILE_DW];             // 2 x 16KB double buffer

    const int lane = threadIdx.x & 31;
    const int wave = threadIdx.x >> 5;
    const int row0 = (blockIdx.x * 8 + wave) * 16;    // 16-row tile base

    const int m    = lane & 15;                       // A-row / B-col of lane
    const int half = lane >> 4;

    // ---- Load + convert the x tile into A fragments (hi/lo), fuse |x|^2 ----
    const float* xrow = x + (size_t)(row0 + m) * DIM;
    v8u ah[8], al[8];
    float xn = 0.0f;
#pragma unroll
    for (int s = 0; s < 8; ++s) {
        float g[16];
        const float4* p0 = reinterpret_cast<const float4*>(xrow + s * 32 + half * 16);
        const float4* p1 = reinterpret_cast<const float4*>(xrow + s * 32 + half * 16 + 8);
        float4 q0 = p0[0], q1 = p0[1], q2 = p1[0], q3 = p1[1];
        g[0]  = q0.x; g[1]  = q0.y; g[2]  = q0.z; g[3]  = q0.w;
        g[4]  = q1.x; g[5]  = q1.y; g[6]  = q1.z; g[7]  = q1.w;
        g[8]  = q2.x; g[9]  = q2.y; g[10] = q2.z; g[11] = q2.w;
        g[12] = q3.x; g[13] = q3.y; g[14] = q3.z; g[15] = q3.w;

        v8u uh, ul;
#pragma unroll
        for (int j = 0; j < 8; ++j) {
            const float f0 = g[2 * j + 0];
            const float f1 = g[2 * j + 1];
            const unsigned h0 = f2bf_rne(f0), h1 = f2bf_rne(f1);
            const unsigned l0 = f2bf_rne(f0 - bf_bits_to_f(h0));
            const unsigned l1 = f2bf_rne(f1 - bf_bits_to_f(h1));
            uh[j] = h0 | (h1 << 16);
            ul[j] = l0 | (l1 << 16);
            xn += f0 * f0 + f1 * f1;
        }
        ah[s] = uh;
        al[s] = ul;
    }
    // lanes m and m+16 jointly hold row m -> full |x_m|^2 on both lanes
    xn += __shfl_xor(xn, 16);

    float best[8];
    int   bidx[8];
#pragma unroll
    for (int i = 0; i < 8; ++i) { best[i] = NEG_INF; bidx[i] = 0; }

    // LDS byte offsets of the two staging buffers (low 32 bits of the generic
    // pointer are the hardware LDS offset per the flat-addressing rules).
    const unsigned loff0 = (unsigned)(uintptr_t)(&ldsb[0][0]);
    const unsigned loff1 = (unsigned)(uintptr_t)(&ldsb[1][0]);

    // ---- Prologue: TDM-prefetch tile 0 (one DMA per block, from wave 0) ----
    if (wave == 0) tdm_load_tile(stage, loff0);

    const int ncol = m;                               // column within tile
    for (int t = 0; t < NTILES; ++t) {
        if (wave == 0) {
            if (t + 1 < NTILES) {
                // Prefetch next tile into the other buffer, then wait until
                // only that prefetch is outstanding (TDM completes in order).
                tdm_load_tile(stage + (size_t)(t + 1) * TILE_DW,
                              ((t + 1) & 1) ? loff1 : loff0);
                __builtin_amdgcn_s_wait_tensorcnt(1);
            } else {
                __builtin_amdgcn_s_wait_tensorcnt(0);
            }
        }
        __syncthreads();                              // tile t visible to all

        const unsigned* buf = &ldsb[t & 1][0];
        const uint4* bhrow = reinterpret_cast<const uint4*>(
            buf + (size_t)ncol * (DIM / 2));
        const uint4* blrow = reinterpret_cast<const uint4*>(
            buf + (TILE_DW / 2) + (size_t)ncol * (DIM / 2));

        v8f acc = {};
#pragma unroll
        for (int s = 0; s < 8; ++s) {
            const int q = s * 4 + half * 2;           // uint4 offset in row
            uint4 h0 = bhrow[q], h1 = bhrow[q + 1];
            uint4 l0 = blrow[q], l1 = blrow[q + 1];
            v8u bh = {h0.x, h0.y, h0.z, h0.w, h1.x, h1.y, h1.z, h1.w};
            v8u bl = {l0.x, l0.y, l0.z, l0.w, l1.x, l1.y, l1.z, l1.w};
            // bf16x3: hi*hi + lo*hi + hi*lo  (lo*lo dropped, below fp32 ulp)
            acc = __builtin_amdgcn_wmma_f32_16x16x32_bf16(
                false, as_v16bf(ah[s]), false, as_v16bf(bh), (short)0, acc, false, false);
            acc = __builtin_amdgcn_wmma_f32_16x16x32_bf16(
                false, as_v16bf(al[s]), false, as_v16bf(bh), (short)0, acc, false, false);
            acc = __builtin_amdgcn_wmma_f32_16x16x32_bf16(
                false, as_v16bf(ah[s]), false, as_v16bf(bl), (short)0, acc, false, false);
        }
        const int   n   = t * 16 + ncol;
        const float cn2 = cnorm[n];
#pragma unroll
        for (int i = 0; i < 8; ++i) {
            const float s2 = 2.0f * acc[i] - cn2;     // |x|^2 is row-constant
            if (s2 > best[i]) { best[i] = s2; bidx[i] = n; }
        }
        __syncthreads();      // all waves done with buffer t&1 before t+2 DMA
    }

    // ---- Cross-lane argmax within each 16-lane group (same rows) ----
#pragma unroll
    for (int off = 8; off >= 1; off >>= 1) {
#pragma unroll
        for (int i = 0; i < 8; ++i) {
            const float ov = __shfl_xor(best[i], off);
            const int   oi = __shfl_xor(bidx[i], off);
            if (ov > best[i] || (ov == best[i] && oi < bidx[i])) {
                best[i] = ov;
                bidx[i] = oi;
            }
        }
    }

    // Lane with L%16 == row (and matching half) owns |x_row|^2 and writes it.
    if ((m >> 3) == half) {
        const int i = m & 7;                 // row m == i + 8*half
        out_sim[row0 + m] = best[i] - xn;
        out_idx[row0 + m] = bidx[i];
    }
}

// ---------------------------------------------------------------------------
// Host-side launcher.
//   d_in[0] = x         fp32 [131072*256]
//   d_in[1] = centroids fp32 [1024*256]
//   d_out   = [N] max_sim (f32) ++ [N] labels (i32 bits)
//   d_ws    >= 64*16KB + 4KB (~1.05 MB)
// ---------------------------------------------------------------------------
extern "C" void kernel_launch(void* const* d_in, const int* in_sizes, int n_in,
                              void* d_out, int out_size, void* d_ws, size_t ws_size,
                              hipStream_t stream) {
    (void)n_in; (void)out_size; (void)ws_size;

    const float* x   = (const float*)d_in[0];
    const float* cen = (const float*)d_in[1];
    const int N = in_sizes[0] / DIM;   // 131072

    char* ws = (char*)d_ws;
    unsigned* stage = (unsigned*)ws;                                  // 1 MB
    float*    cnm   = (float*)(ws + (size_t)NTILES * TILE_DW * 4);    // 4 KB

    float* out_sim = (float*)d_out;
    int*   out_idx = (int*)d_out + N;

    centroid_prep_kernel<<<KC / 8, 256, 0, stream>>>(cen, stage, cnm);
    nearest_centroid_kernel<<<N / 16 / 8, 256, 0, stream>>>(
        x, stage, cnm, out_sim, out_idx);
}
